// GIN_32512902431423
// MI455X (gfx1250) — matrix operand order repaired and checked by reading, not compiled, not run on hardware
//
#include <hip/hip_runtime.h>

// ---------------------------------------------------------------------------
// GIN forward on MI455X (gfx1250, wave32).
// Scatter: f32 atomics (L2-resident). GEMMs: v_wmma_f32_16x16x32_bf16.
// ---------------------------------------------------------------------------

typedef __attribute__((ext_vector_type(16))) __bf16 bf16x16;
typedef __attribute__((ext_vector_type(8)))  __bf16 bf16x8;
typedef __attribute__((ext_vector_type(8)))  float  f32x8;

struct alignas(8) bh4 { __bf16 v[4]; };

// ---------------- weight transpose + f32->bf16 convert (Wt[m][k] = W[k][m])
__global__ void wt_convert_kernel(const float* __restrict__ W, __bf16* __restrict__ Wt,
                                  int K, int M, int Mpad) {
    int idx = blockIdx.x * blockDim.x + threadIdx.x;
    if (idx >= K * Mpad) return;
    int m = idx / K;
    int k = idx - m * K;
    Wt[(size_t)m * K + k] = (m < M) ? (__bf16)W[(size_t)k * M + m] : (__bf16)0.0f;
}

// ---------------- f32 vector copy (z1 = x init)
__global__ void copy_f32x4_kernel(const float* __restrict__ in, float* __restrict__ out,
                                  size_t n4) {
    size_t idx = (size_t)blockIdx.x * blockDim.x + threadIdx.x;
    if (idx < n4) ((float4*)out)[idx] = ((const float4*)in)[idx];
}

// ---------------- f32 -> bf16 convert (activation staging for WMMA A operand)
__global__ void f32_to_bf16_kernel(const float* __restrict__ in, __bf16* __restrict__ out,
                                   size_t n4) {
    size_t idx = (size_t)blockIdx.x * blockDim.x + threadIdx.x;
    if (idx >= n4) return;
    float4 v = ((const float4*)in)[idx];
    bh4 o;
    o.v[0] = (__bf16)v.x; o.v[1] = (__bf16)v.y;
    o.v[2] = (__bf16)v.z; o.v[3] = (__bf16)v.w;
    ((bh4*)out)[idx] = o;
}

// ---------------- edge scatter-add: accum[dst] += feat[src]  (float4 gather,
// 4x global_atomic_add_f32; targets fit in 192MB L2 so atomics stay on-chip)
__global__ void scatter_add_kernel(const float* __restrict__ feat,
                                   float* __restrict__ accum,
                                   const int* __restrict__ edge_index,
                                   int nEdges, int featDim) {
    int t = blockIdx.x * blockDim.x + threadIdx.x;
    int chunks = featDim >> 2;                 // float4 chunks per edge
    int e = t / chunks;
    int c = t - e * chunks;
    if (e >= nEdges) return;
    int s = edge_index[e];                     // row 0: src
    int d = edge_index[nEdges + e];            // row 1: dst
    float4 v = *(const float4*)(feat + (size_t)s * featDim + (size_t)c * 4);
    float* p = accum + (size_t)d * featDim + (size_t)c * 4;
    atomicAdd(p + 0, v.x);
    atomicAdd(p + 1, v.y);
    atomicAdd(p + 2, v.z);
    atomicAdd(p + 3, v.w);
}

// ---------------- bf16 WMMA GEMM: D[16x16 per wave] = A[N,K] * Wt[M,K]^T + bias
// OUTMODE: 0 = bf16 out (relu-staged activation), 1 = f32 dual write (h and z2),
//          2 = f32 masked columns (final classifier, ncols=10)
template <int K, int OUTMODE, bool RELU>
__global__ __launch_bounds__(512)
void gemm_wmma_kernel(const __bf16* __restrict__ A,
                      const __bf16* __restrict__ Wt,
                      const float* __restrict__ bias,
                      __bf16* __restrict__ outb,
                      float* __restrict__ outf0,
                      float* __restrict__ outf1,
                      int ldo, int ncols) {
    const int lane = threadIdx.x & 31;
    const int wave = threadIdx.x >> 5;
    const int row0 = blockIdx.x * 16;
    const int col0 = wave * 16;
    const int r    = lane & 15;
    const int hi   = lane >> 4;           // lane half-select (ISA 7.12.2 layouts)

    const __bf16* Arow = A  + (size_t)(row0 + r) * K;   // lane's A row (M = lane%16)
    const __bf16* Brow = Wt + (size_t)(col0 + r) * K;   // lane's B col (N = lane%16)

    f32x8 acc = {0.f, 0.f, 0.f, 0.f, 0.f, 0.f, 0.f, 0.f};

#pragma unroll
    for (int kk = 0; kk < K; kk += 32) {
        // A 16x32 bf16 fragment: lanes 0-15 hold K={kk..kk+7, kk+16..kk+23},
        // lanes 16-31 hold K={kk+8..kk+15, kk+24..kk+31}
        bf16x8 a_lo = *(const bf16x8*)(Arow + kk + 8 * hi);
        bf16x8 a_hi = *(const bf16x8*)(Arow + kk + 16 + 8 * hi);
        // B 32x16 bf16 fragment: lanes 0-15 hold K={kk..kk+7, kk+16..kk+23},
        // lanes 16-31 hold K={kk+8..kk+15, kk+24..kk+31}
        bf16x8 b_lo = *(const bf16x8*)(Brow + kk + 8 * hi);
        bf16x8 b_hi = *(const bf16x8*)(Brow + kk + 16 + 8 * hi);
        bf16x16 a, b;
#pragma unroll
        for (int i = 0; i < 8; ++i) {
            a[i] = a_lo[i]; a[i + 8] = a_hi[i];
            b[i] = b_lo[i]; b[i + 8] = b_hi[i];
        }
        acc = __builtin_amdgcn_wmma_f32_16x16x32_bf16(
            /*neg_a=*/false, a, /*neg_b=*/false, b,
            /*c_mod=*/(short)0, acc, /*reuse_a=*/false, /*reuse_b=*/false);
    }

    const int ocol = col0 + r;
    float bv = (ocol < ncols) ? bias[ocol] : 0.0f;

#pragma unroll
    for (int i = 0; i < 8; ++i) {
        int orow = row0 + i + 8 * hi;     // D: VGPR i, lanes 0-15 -> M=i, 16-31 -> M=i+8
        float v = acc[i] + bv;
        if (RELU) v = fmaxf(v, 0.0f);
        if (OUTMODE == 0) {
            outb[(size_t)orow * ldo + ocol] = (__bf16)v;
        } else if (OUTMODE == 1) {
            outf0[(size_t)orow * ldo + ocol] = v;
            outf1[(size_t)orow * ldo + ocol] = v;
        } else {
            if (ocol < ncols) outf0[(size_t)orow * ldo + ocol] = v;
        }
    }
}

// ---------------------------------------------------------------------------
extern "C" void kernel_launch(void* const* d_in, const int* in_sizes, int n_in,
                              void* d_out, int out_size, void* d_ws, size_t ws_size,
                              hipStream_t stream) {
    constexpr int F = 128, H = 256, C = 10;
    const int N = in_sizes[0] / F;     // 100000
    const int E = in_sizes[1] / 2;     // 640000

    const float* x   = (const float*)d_in[0];
    const int*   ei  = (const int*)  d_in[1];
    const float* W1a = (const float*)d_in[2];
    const float* b1a = (const float*)d_in[3];
    const float* W1b = (const float*)d_in[4];
    const float* b1b = (const float*)d_in[5];
    const float* W2a = (const float*)d_in[6];
    const float* b2a = (const float*)d_in[7];
    const float* W2b = (const float*)d_in[8];
    const float* b2b = (const float*)d_in[9];
    float* out = (float*)d_out;

    // ---- workspace carve-up (256B aligned)
    char* base = (char*)d_ws;
    size_t off = 0;
    auto carve = [&](size_t bytes) -> void* {
        void* p = base + off;
        off = (off + bytes + 255) & ~(size_t)255;
        return p;
    };
    float*  z1   = (float*) carve((size_t)N * F * 4);   // x + agg1
    __bf16* zb1  = (__bf16*)carve((size_t)N * F * 2);   // bf16(z1)
    __bf16* t1b  = (__bf16*)carve((size_t)N * H * 2);   // relu(z1 W1a + b1a)
    float*  hbuf = (float*) carve((size_t)N * H * 4);   // relu(t1 W1b + b1b)
    float*  z2   = (float*) carve((size_t)N * H * 4);   // h + agg2
    __bf16* zb2  = (__bf16*)carve((size_t)N * H * 2);   // bf16(z2)
    __bf16* t2b  = (__bf16*)carve((size_t)N * H * 2);   // relu(z2 W2a + b2a)
    __bf16* w1aT = (__bf16*)carve((size_t)H * F * 2);   // [256][128]
    __bf16* w1bT = (__bf16*)carve((size_t)H * H * 2);   // [256][256]
    __bf16* w2aT = (__bf16*)carve((size_t)H * H * 2);   // [256][256]
    __bf16* w2bT = (__bf16*)carve((size_t)16 * H * 2);  // [16][256], rows 10..15 zero
    (void)ws_size;

    // ---- weight prep: transpose + bf16
    wt_convert_kernel<<<(F * H + 255) / 256, 256, 0, stream>>>(W1a, w1aT, F, H, H);
    wt_convert_kernel<<<(H * H + 255) / 256, 256, 0, stream>>>(W1b, w1bT, H, H, H);
    wt_convert_kernel<<<(H * H + 255) / 256, 256, 0, stream>>>(W2a, w2aT, H, H, H);
    wt_convert_kernel<<<(H * 16 + 255) / 256, 256, 0, stream>>>(W2b, w2bT, H, C, 16);

    // ---- layer 1: z1 = x + segment_sum(x[src] -> dst)
    {
        size_t n4 = (size_t)N * F / 4;
        copy_f32x4_kernel<<<(unsigned)((n4 + 255) / 256), 256, 0, stream>>>(x, z1, n4);
        size_t nt = (size_t)E * (F / 4);
        scatter_add_kernel<<<(unsigned)((nt + 255) / 256), 256, 0, stream>>>(x, z1, ei, E, F);
        f32_to_bf16_kernel<<<(unsigned)((n4 + 255) / 256), 256, 0, stream>>>(z1, zb1, n4);
    }

    // ---- t1 = relu(z1 @ W1a + b1a)  -> bf16
    gemm_wmma_kernel<F, 0, true><<<N / 16, 512, 0, stream>>>(
        zb1, w1aT, b1a, t1b, nullptr, nullptr, H, H);

    // ---- h = relu(t1 @ W1b + b1b) -> f32, dual-written to hbuf and z2 (fused init)
    gemm_wmma_kernel<H, 1, true><<<N / 16, 512, 0, stream>>>(
        t1b, w1bT, b1b, nullptr, hbuf, z2, H, H);

    // ---- layer 2 aggregation: z2 += segment_sum(h[src] -> dst)
    {
        size_t nt = (size_t)E * (H / 4);
        scatter_add_kernel<<<(unsigned)((nt + 255) / 256), 256, 0, stream>>>(hbuf, z2, ei, E, H);
        size_t n4 = (size_t)N * H / 4;
        f32_to_bf16_kernel<<<(unsigned)((n4 + 255) / 256), 256, 0, stream>>>(z2, zb2, n4);
    }

    // ---- t2 = relu(z2 @ W2a + b2a) -> bf16
    gemm_wmma_kernel<H, 0, true><<<N / 16, 512, 0, stream>>>(
        zb2, w2aT, b2a, t2b, nullptr, nullptr, H, H);

    // ---- out = t2 @ W2b + b2b -> f32 [N,10] (1 wave/block, masked col stores)
    gemm_wmma_kernel<H, 2, false><<<N / 16, 32, 0, stream>>>(
        t2b, w2bT, b2b, nullptr, out, nullptr, C, C);
}
